// TransformerBlock_38628935860422
// MI455X (gfx1250) — matrix operand order repaired and checked by reading, not compiled, not run on hardware
//
#include <hip/hip_runtime.h>
#include <hip/hip_bf16.h>
#include <math.h>

// Problem constants
#define B_    2
#define T_    2048
#define D_    1024
#define H_    16
#define DK_   64
#define DFF_  2730
#define DFFP_ 2752          // padded to multiple of 64 for WMMA tiling
#define ROWS_ (B_*T_)       // 4096

typedef __attribute__((ext_vector_type(16))) __bf16 v16bf;
typedef __attribute__((ext_vector_type(8)))  float  v8f;
typedef __attribute__((ext_vector_type(4)))  unsigned int v4ui;

union AB16 {
    v16bf v;
    unsigned short h[16];
    unsigned int   w[8];
    v4ui           q[2];
};

__device__ __forceinline__ unsigned short f2bf(float f) {
    unsigned int u = __float_as_uint(f);
    unsigned int r = (u + 0x7FFFu + ((u >> 16) & 1u)) >> 16;  // RNE
    return (unsigned short)r;
}
__device__ __forceinline__ float bf2f(unsigned short h) {
    return __uint_as_float(((unsigned int)h) << 16);
}

__device__ __forceinline__ v8f wmma_bf16(const AB16& a, const AB16& b, v8f c) {
    return __builtin_amdgcn_wmma_f32_16x16x32_bf16(false, a.v, false, b.v,
                                                   (short)0, c, false, false);
}

__device__ __forceinline__ float hmax16(float x) {
    x = fmaxf(x, __shfl_xor(x, 1, 32));
    x = fmaxf(x, __shfl_xor(x, 2, 32));
    x = fmaxf(x, __shfl_xor(x, 4, 32));
    x = fmaxf(x, __shfl_xor(x, 8, 32));
    return x;
}
__device__ __forceinline__ float hsum16(float x) {
    x += __shfl_xor(x, 1, 32);
    x += __shfl_xor(x, 2, 32);
    x += __shfl_xor(x, 4, 32);
    x += __shfl_xor(x, 8, 32);
    return x;
}

// ---- weight fp32 [N,K] -> bf16 TRANSPOSED [K,Npad/Kpad] with zero pad ----
__global__ void k_cvt_tpad(const float* __restrict__ src, unsigned short* __restrict__ dst,
                           int dRows, int dCols, int sRows, int sCols) {
    long tot = (long)dRows * dCols;
    for (long idx = (long)blockIdx.x * blockDim.x + threadIdx.x; idx < tot;
         idx += (long)gridDim.x * blockDim.x) {
        int kr = (int)(idx / dCols), nc = (int)(idx % dCols);
        float f = (nc < sRows && kr < sCols) ? src[(long)nc * sCols + kr] : 0.0f;
        dst[idx] = f2bf(f);
    }
}

// ---------------- RMSNorm: fp32 row -> bf16 row ---------------------------
__global__ __launch_bounds__(256) void k_rmsnorm(const float* __restrict__ x,
                                                 const float* __restrict__ g,
                                                 unsigned short* __restrict__ out) {
    int row = blockIdx.x;
    const float* xr = x + (long)row * D_;
    __shared__ float red[256];
    float vals[4];
    float s = 0.0f;
#pragma unroll
    for (int j = 0; j < 4; ++j) {
        vals[j] = xr[threadIdx.x * 4 + j];
        s += vals[j] * vals[j];
    }
    red[threadIdx.x] = s;
    __syncthreads();
    for (int off = 128; off > 0; off >>= 1) {
        if (threadIdx.x < off) red[threadIdx.x] += red[threadIdx.x + off];
        __syncthreads();
    }
    float inv = rsqrtf(red[0] * (1.0f / D_) + 1e-5f);
#pragma unroll
    for (int j = 0; j < 4; ++j) {
        int c = threadIdx.x * 4 + j;
        out[(long)row * D_ + c] = f2bf(vals[j] * inv * g[c]);
    }
}

// ---------------- NT GEMM with pre-transposed weights ---------------------
// C[M,N] = A[M,K](bf16) * Wt[K,N](bf16)
// Block tile 256(M) x 64(N); 8 waves each own 32 rows x 64 cols (8 accs).
// Wt k-slab (32x64) staged via async-to-LDS, double buffered, 1 barrier/step.
// EPI==0: store bf16.  EPI==1: outF = res + acc (fp32); res may alias outF.
template <int EPI>
__global__ __launch_bounds__(256) void k_gemm(const unsigned short* __restrict__ A,
                                              const unsigned short* __restrict__ Wt,
                                              const float* res,
                                              unsigned short* __restrict__ outBf,
                                              float* outF,
                                              int M, int N, int K) {
    __shared__ __align__(16) unsigned short shW[2][32][72];  // 32k x 64n (+8 pad)
    int tid  = threadIdx.x;
    int lane = tid & 31;
    int w    = tid >> 5;
    int m0   = blockIdx.y * 256 + w * 32;
    int n0   = blockIdx.x * 64;
    int mrow = lane & 15, hi = lane >> 4;
    int kt   = tid >> 3;          // 0..31 : k row of slab this thread copies
    int c8   = (tid & 7) * 8;     // 0..56 : n-chunk (8 elements = 16B)

    unsigned ldsDst0 = (unsigned)(size_t)&shW[0][kt][c8];
    unsigned ldsDst1 = (unsigned)(size_t)&shW[1][kt][c8];

    v8f acc[2][4] = {};
    const unsigned short* Arow0 = A + (long)(m0 + mrow) * K;
    const unsigned short* Arow1 = Arow0 + (long)16 * K;

    int nk = K >> 5;
    {   // prologue: async-stage slab 0 into buffer 0
        const unsigned short* gsrc = Wt + (long)kt * N + n0 + c8;
        asm volatile("global_load_async_to_lds_b128 %0, %1, off"
                     :: "v"(ldsDst0), "v"((unsigned long long)(size_t)gsrc) : "memory");
    }
    for (int ik = 0; ik < nk; ++ik) {
        int kb  = ik << 5;
        int cur = ik & 1;
        asm volatile("s_wait_asynccnt 0" ::: "memory");
        __syncthreads();                       // slab `cur` visible; prev reads done
        if (ik + 1 < nk) {                     // stream next slab into other buffer
            const unsigned short* gsrc = Wt + (long)(kb + 32 + kt) * N + n0 + c8;
            asm volatile("global_load_async_to_lds_b128 %0, %1, off"
                         :: "v"(cur ? ldsDst0 : ldsDst1),
                            "v"((unsigned long long)(size_t)gsrc) : "memory");
        }
        // A operands: two row-halves, two contiguous 16B loads each
        AB16 a0, a1;
        a0.q[0] = *(const v4ui*)(Arow0 + kb + 8 * hi);
        a0.q[1] = *(const v4ui*)(Arow0 + kb + 16 + 8 * hi);
        a1.q[0] = *(const v4ui*)(Arow1 + kb + 8 * hi);
        a1.q[1] = *(const v4ui*)(Arow1 + kb + 16 + 8 * hi);
        __builtin_prefetch(Arow0 + kb + 32, 0, 1);
        __builtin_prefetch(Arow1 + kb + 32, 0, 1);
        // B operands from LDS: lane = n column, 16 k (table order) per tile
        AB16 b0, b1, b2, b3;
        const unsigned short* sr = &shW[cur][0][lane & 15];
#pragma unroll
        for (int s = 0; s < 16; ++s) {
            int kk = (s & 7) + 8 * hi + 16 * (s >> 3);
            b0.h[s] = sr[kk * 72 +  0]; b1.h[s] = sr[kk * 72 + 16];
            b2.h[s] = sr[kk * 72 + 32]; b3.h[s] = sr[kk * 72 + 48];
        }
        acc[0][0] = wmma_bf16(a0, b0, acc[0][0]);
        acc[1][0] = wmma_bf16(a1, b0, acc[1][0]);
        acc[0][1] = wmma_bf16(a0, b1, acc[0][1]);
        acc[1][1] = wmma_bf16(a1, b1, acc[1][1]);
        acc[0][2] = wmma_bf16(a0, b2, acc[0][2]);
        acc[1][2] = wmma_bf16(a1, b2, acc[1][2]);
        acc[0][3] = wmma_bf16(a0, b3, acc[0][3]);
        acc[1][3] = wmma_bf16(a1, b3, acc[1][3]);
    }

    int colb = lane & 15;
#pragma unroll
    for (int half = 0; half < 2; ++half) {
#pragma unroll
        for (int r = 0; r < 8; ++r) {
            int row = m0 + 16 * half + r + 8 * hi;
            long base = (long)row * N + n0 + colb;
            if (EPI == 0) {
                outBf[base]      = f2bf(acc[half][0][r]);
                outBf[base + 16] = f2bf(acc[half][1][r]);
                outBf[base + 32] = f2bf(acc[half][2][r]);
                outBf[base + 48] = f2bf(acc[half][3][r]);
            } else {
                outF[base]      = res[base]      + acc[half][0][r];
                outF[base + 16] = res[base + 16] + acc[half][1][r];
                outF[base + 32] = res[base + 32] + acc[half][2][r];
                outF[base + 48] = res[base + 48] + acc[half][3][r];
            }
        }
    }
}

// -------- RoPE on q (in place) + fold in softmax scale 1/sqrt(dk) ---------
__global__ void k_rope_q(unsigned short* __restrict__ qk, const int* __restrict__ pos) {
    long tot = (long)ROWS_ * (D_ / 2);
    for (long idx = (long)blockIdx.x * blockDim.x + threadIdx.x; idx < tot;
         idx += (long)gridDim.x * blockDim.x) {
        int row = (int)(idx / (D_ / 2));
        int pr  = (int)(idx % (D_ / 2));
        int hh  = pr / (DK_ / 2);
        int i   = pr % (DK_ / 2);
        float p = (float)pos[row];
        float ang = p * powf(10000.0f, -(float)i / (DK_ / 2));
        float s, c;
        sincosf(ang, &s, &c);
        long e = (long)row * D_ + hh * DK_ + 2 * i;
        float x0 = bf2f(qk[e]), x1 = bf2f(qk[e + 1]);
        qk[e]     = f2bf(0.125f * (c * x0 - s * x1));   // 1/8 exact in bf16
        qk[e + 1] = f2bf(0.125f * (s * x0 + c * x1));
    }
}

// ------- RoPE on k + transpose into kT[(b*H+h)*64 + d][t] -----------------
__global__ void k_rope_t(const unsigned short* __restrict__ kin,
                         unsigned short* __restrict__ kT,
                         const int* __restrict__ pos) {
    long tot = (long)B_ * H_ * (DK_ / 2) * T_;   // t fastest -> coalesced writes
    for (long idx = (long)blockIdx.x * blockDim.x + threadIdx.x; idx < tot;
         idx += (long)gridDim.x * blockDim.x) {
        int  t  = (int)(idx % T_);
        long r  = idx / T_;
        int  i  = (int)(r % (DK_ / 2));
        int  bh = (int)(r / (DK_ / 2));          // 0..31
        int  b  = bh / H_, hh = bh % H_;
        int  row = b * T_ + t;
        float p = (float)pos[row];
        float ang = p * powf(10000.0f, -(float)i / (DK_ / 2));
        float s, c;
        sincosf(ang, &s, &c);
        long e = (long)row * D_ + hh * DK_ + 2 * i;
        float x0 = bf2f(kin[e]), x1 = bf2f(kin[e + 1]);
        kT[((long)bh * DK_ + 2 * i)     * T_ + t] = f2bf(c * x0 - s * x1);
        kT[((long)bh * DK_ + 2 * i + 1) * T_ + t] = f2bf(s * x0 + c * x1);
    }
}

// ---------------- Flash attention (causal) --------------------------------
// One wave handles 16 query rows of one (batch, head); kv tiles of 64 keys
// (4 score tiles) so the per-row softmax butterflies amortize over 2x keys.
// All wave-uniform control values go through readfirstlane -> SALU branches,
// keeping EXEC all-ones at every WMMA.
__global__ __launch_bounds__(256) void k_attn(const unsigned short* __restrict__ q,
                                              const unsigned short* __restrict__ kT,
                                              const unsigned short* __restrict__ v,
                                              unsigned short* __restrict__ o) {
    __shared__ __align__(16) unsigned short psh[8][16 * 64];  // per-wave P staging (2KB)
    int lane = threadIdx.x & 31;
    int w    = threadIdx.x >> 5;
    int gw   = __builtin_amdgcn_readfirstlane(blockIdx.x * 8 + (threadIdx.x >> 5));
    int b    = gw / (H_ * (T_ / 16));
    int rem  = gw % (H_ * (T_ / 16));
    int hh   = rem / (T_ / 16);
    int qt   = rem % (T_ / 16);
    int qw   = qt * 16;                       // uniform (SGPR)
    int hb   = hh * DK_;
    long rowbase = (long)b * T_ * D_;
    int mrow = lane & 15, hi = lane >> 4;
    int col  = lane & 15;

    // Q as two A operands (d = 0..31, 32..63): 4 x 16B loads
    AB16 a0, a1;
    const unsigned short* Qrow = q + rowbase + (long)(qw + mrow) * D_ + hb;
    a0.q[0] = *(const v4ui*)(Qrow + 8 * hi);
    a0.q[1] = *(const v4ui*)(Qrow + 16 + 8 * hi);
    a1.q[0] = *(const v4ui*)(Qrow + 32 + 8 * hi);
    a1.q[1] = *(const v4ui*)(Qrow + 48 + 8 * hi);

    // K^T column for this lane's key (chunk0: d=0..31, chunk1: d=32..63)
    const unsigned short* KtL = kT + (long)(b * H_ + hh) * DK_ * T_ + col;
    const unsigned short* KtH = KtL + (long)32 * T_;

    v8f oa0 = {}, oa1 = {}, oa2 = {}, oa3 = {};
    float mrw[8], lrw[8];
#pragma unroll
    for (int r = 0; r < 8; ++r) { mrw[r] = -INFINITY; lrw[r] = 0.0f; }

    for (int j0 = 0; j0 <= qw + 15; j0 += 64) {
        // ---- S = Q * K^T : four 16-key tiles, contiguous 32B loads ----
        v8f s0 = {}, s1 = {}, s2 = {}, s3 = {};
        {
            AB16 bA[4], bB[4];
#pragma unroll
            for (int s = 0; s < 16; ++s) {
                long kk = (s & 7) + 8 * hi + 16 * (s >> 3);
#pragma unroll
                for (int t = 0; t < 4; ++t) {
                    bA[t].h[s] = KtL[kk * T_ + j0 + 16 * t];
                    bB[t].h[s] = KtH[kk * T_ + j0 + 16 * t];
                }
            }
            s0 = wmma_bf16(a0, bA[0], s0);         s0 = wmma_bf16(a1, bB[0], s0);
            s1 = wmma_bf16(a0, bA[1], s1);         s1 = wmma_bf16(a1, bB[1], s1);
            s2 = wmma_bf16(a0, bA[2], s2);         s2 = wmma_bf16(a1, bB[2], s2);
            s3 = wmma_bf16(a0, bA[3], s3);         s3 = wmma_bf16(a1, bB[3], s3);
        }

        // ---- causal mask: only the trailing tiles need it (SALU branch) ----
        if (j0 + 63 > qw) {
            int key = j0 + col;
#pragma unroll
            for (int r = 0; r < 8; ++r) {
                int qr = qw + r + 8 * hi;
                s0[r] = (key      <= qr) ? s0[r] : -INFINITY;
                s1[r] = (key + 16 <= qr) ? s1[r] : -INFINITY;
                s2[r] = (key + 32 <= qr) ? s2[r] : -INFINITY;
                s3[r] = (key + 48 <= qr) ? s3[r] : -INFINITY;
            }
        }

        // ---- online softmax update (exp in place) ----
#pragma unroll
        for (int r = 0; r < 8; ++r) {
            float tmax = fmaxf(fmaxf(s0[r], s1[r]), fmaxf(s2[r], s3[r]));
            float rmax = hmax16(tmax);
            float mnew = fmaxf(mrw[r], rmax);
            float alpha = __expf(mrw[r] - mnew);
            s0[r] = __expf(s0[r] - mnew);
            s1[r] = __expf(s1[r] - mnew);
            s2[r] = __expf(s2[r] - mnew);
            s3[r] = __expf(s3[r] - mnew);
            float rsum = hsum16((s0[r] + s1[r]) + (s2[r] + s3[r]));
            lrw[r] = lrw[r] * alpha + rsum;
            mrw[r] = mnew;
            oa0[r] *= alpha; oa1[r] *= alpha; oa2[r] *= alpha; oa3[r] *= alpha;
        }

        // ---- C-layout -> A-layout for P via wave-private LDS ----
#pragma unroll
        for (int r = 0; r < 8; ++r) {
            int rr = (r + 8 * hi) * 64 + col;
            psh[w][rr]      = f2bf(s0[r]);
            psh[w][rr + 16] = f2bf(s1[r]);
            psh[w][rr + 32] = f2bf(s2[r]);
            psh[w][rr + 48] = f2bf(s3[r]);
        }
        asm volatile("s_wait_dscnt 0" ::: "memory");
        AB16 pa0, pa1;   // keys j0..j0+31 and j0+32..j0+63
        pa0.q[0] = *(const v4ui*)&psh[w][mrow * 64 + 8 * hi];
        pa0.q[1] = *(const v4ui*)&psh[w][mrow * 64 + 16 + 8 * hi];
        pa1.q[0] = *(const v4ui*)&psh[w][mrow * 64 + 32 + 8 * hi];
        pa1.q[1] = *(const v4ui*)&psh[w][mrow * 64 + 48 + 8 * hi];

        // ---- O += P(16x64) * V(64x64): V B-operands, lane = d, elements = keys ----
        const unsigned short* Vb0 = v + rowbase + (long)j0 * D_ + hb + col;
        {
            AB16 bv0, bv1, bv2, bv3;
#pragma unroll
            for (int s = 0; s < 16; ++s) {
                long kk = (s & 7) + 8 * hi + 16 * (s >> 3);
                bv0.h[s] = Vb0[kk * D_];      bv1.h[s] = Vb0[kk * D_ + 16];
                bv2.h[s] = Vb0[kk * D_ + 32]; bv3.h[s] = Vb0[kk * D_ + 48];
            }
            oa0 = wmma_bf16(pa0, bv0, oa0);
            oa1 = wmma_bf16(pa0, bv1, oa1);
            oa2 = wmma_bf16(pa0, bv2, oa2);
            oa3 = wmma_bf16(pa0, bv3, oa3);
        }
        const unsigned short* Vb1 = Vb0 + (long)32 * D_;
        {
            AB16 bv0, bv1, bv2, bv3;
#pragma unroll
            for (int s = 0; s < 16; ++s) {
                long kk = (s & 7) + 8 * hi + 16 * (s >> 3);
                bv0.h[s] = Vb1[kk * D_];      bv1.h[s] = Vb1[kk * D_ + 16];
                bv2.h[s] = Vb1[kk * D_ + 32]; bv3.h[s] = Vb1[kk * D_ + 48];
            }
            oa0 = wmma_bf16(pa1, bv0, oa0);
            oa1 = wmma_bf16(pa1, bv1, oa1);
            oa2 = wmma_bf16(pa1, bv2, oa2);
            oa3 = wmma_bf16(pa1, bv3, oa3);
        }
    }

    // ---- normalize and write bf16 O ----
#pragma unroll
    for (int r = 0; r < 8; ++r) {
        int qr = qw + r + 8 * hi;
        float invl = 1.0f / lrw[r];
        long base = rowbase + (long)qr * D_ + hb + col;
        o[base]      = f2bf(oa0[r] * invl);
        o[base + 16] = f2bf(oa1[r] * invl);
        o[base + 32] = f2bf(oa2[r] * invl);
        o[base + 48] = f2bf(oa3[r] * invl);
    }
}

// ---------------- SwiGLU elementwise: u <- silu(u) * g --------------------
__global__ void k_silu(unsigned short* __restrict__ u, const unsigned short* __restrict__ g) {
    long tot = (long)ROWS_ * DFFP_;
    for (long idx = (long)blockIdx.x * blockDim.x + threadIdx.x; idx < tot;
         idx += (long)gridDim.x * blockDim.x) {
        float uf = bf2f(u[idx]);
        float gf = bf2f(g[idx]);
        float s  = uf / (1.0f + __expf(-uf));
        u[idx] = f2bf(s * gf);
    }
}

// ==========================================================================
extern "C" void kernel_launch(void* const* d_in, const int* in_sizes, int n_in,
                              void* d_out, int out_size, void* d_ws, size_t ws_size,
                              hipStream_t stream) {
    const float* x   = (const float*)d_in[0];
    const int*   pos = (const int*)d_in[1];
    const float* WQ  = (const float*)d_in[2];
    const float* WK  = (const float*)d_in[3];
    const float* WV  = (const float*)d_in[4];
    const float* WO  = (const float*)d_in[5];
    const float* W1  = (const float*)d_in[6];
    const float* W2  = (const float*)d_in[7];
    const float* W3  = (const float*)d_in[8];
    const float* g1  = (const float*)d_in[9];
    const float* g2  = (const float*)d_in[10];
    float* out = (float*)d_out;

    char* ws = (char*)d_ws;
    size_t off = 0;
    auto alloc = [&](size_t bytes) -> void* {
        void* p = ws + off;
        off += (bytes + 255) & ~(size_t)255;
        return p;
    };
    // transposed bf16 weights: Wt[K][N]
    unsigned short* WQt = (unsigned short*)alloc((size_t)D_ * D_ * 2);
    unsigned short* WKt = (unsigned short*)alloc((size_t)D_ * D_ * 2);
    unsigned short* WVt = (unsigned short*)alloc((size_t)D_ * D_ * 2);
    unsigned short* WOt = (unsigned short*)alloc((size_t)D_ * D_ * 2);
    unsigned short* W1t = (unsigned short*)alloc((size_t)D_ * DFFP_ * 2);   // [1024][2752]
    unsigned short* W3t = (unsigned short*)alloc((size_t)D_ * DFFP_ * 2);   // [1024][2752]
    unsigned short* W2t = (unsigned short*)alloc((size_t)DFFP_ * D_ * 2);   // [2752][1024]
    unsigned short* hb  = (unsigned short*)alloc((size_t)ROWS_ * D_ * 2);   // h1, reused as h2
    unsigned short* qb  = (unsigned short*)alloc((size_t)ROWS_ * D_ * 2);
    unsigned short* kb  = (unsigned short*)alloc((size_t)ROWS_ * D_ * 2);
    unsigned short* kTb = (unsigned short*)alloc((size_t)ROWS_ * D_ * 2);
    unsigned short* vb  = (unsigned short*)alloc((size_t)ROWS_ * D_ * 2);
    unsigned short* ob  = (unsigned short*)alloc((size_t)ROWS_ * D_ * 2);
    unsigned short* ub  = (unsigned short*)alloc((size_t)ROWS_ * DFFP_ * 2);
    unsigned short* gb  = (unsigned short*)alloc((size_t)ROWS_ * DFFP_ * 2);
    (void)ws_size; (void)in_sizes; (void)n_in; (void)out_size;

    // 1) weights -> transposed bf16 (zero padding of D_FF dimension)
    k_cvt_tpad<<<2048, 256, 0, stream>>>(WQ, WQt, D_, D_, D_, D_);
    k_cvt_tpad<<<2048, 256, 0, stream>>>(WK, WKt, D_, D_, D_, D_);
    k_cvt_tpad<<<2048, 256, 0, stream>>>(WV, WVt, D_, D_, D_, D_);
    k_cvt_tpad<<<2048, 256, 0, stream>>>(WO, WOt, D_, D_, D_, D_);
    k_cvt_tpad<<<2048, 256, 0, stream>>>(W1, W1t, D_, DFFP_, DFF_, D_);
    k_cvt_tpad<<<2048, 256, 0, stream>>>(W3, W3t, D_, DFFP_, DFF_, D_);
    k_cvt_tpad<<<2048, 256, 0, stream>>>(W2, W2t, DFFP_, D_, D_, DFF_);

    // 2) h1 = rmsnorm(x, g1)
    k_rmsnorm<<<ROWS_, 256, 0, stream>>>(x, g1, hb);

    // 3) Q/K/V projections (bf16 out, [b*T, H*DK])
    dim3 gQKV(D_ / 64, ROWS_ / 256);
    k_gemm<0><<<gQKV, 256, 0, stream>>>(hb, WQt, nullptr, qb, nullptr, ROWS_, D_, D_);
    k_gemm<0><<<gQKV, 256, 0, stream>>>(hb, WKt, nullptr, kb, nullptr, ROWS_, D_, D_);
    k_gemm<0><<<gQKV, 256, 0, stream>>>(hb, WVt, nullptr, vb, nullptr, ROWS_, D_, D_);

    // 4) RoPE: q in place (pre-scaled by 1/8); k rotated + transposed
    k_rope_q<<<4096, 256, 0, stream>>>(qb, pos);
    k_rope_t<<<4096, 256, 0, stream>>>(kb, kTb, pos);

    // 5) causal flash attention -> ob
    k_attn<<<(B_ * H_ * (T_ / 16)) / 8, 256, 0, stream>>>(qb, kTb, vb, ob);

    // 6) x1 = x + attn_out * WO^T  (fp32 into d_out)
    k_gemm<1><<<gQKV, 256, 0, stream>>>(ob, WOt, x, nullptr, out, ROWS_, D_, D_);

    // 7) h2 = rmsnorm(x1, g2)
    k_rmsnorm<<<ROWS_, 256, 0, stream>>>(out, g2, hb);

    // 8) u = h2*W1^T, g = h2*W3^T  (padded N = 2752)
    dim3 gFF(DFFP_ / 64, ROWS_ / 256);
    k_gemm<0><<<gFF, 256, 0, stream>>>(hb, W1t, nullptr, ub, nullptr, ROWS_, DFFP_, D_);
    k_gemm<0><<<gFF, 256, 0, stream>>>(hb, W3t, nullptr, gb, nullptr, ROWS_, DFFP_, D_);

    // 9) u <- silu(u) * g
    k_silu<<<8192, 256, 0, stream>>>(ub, gb);

    // 10) out = x1 + (silu(u)*g) * W2^T   (K = 2752, padded tail is zero)
    k_gemm<1><<<gQKV, 256, 0, stream>>>(ub, W2t, out, nullptr, out, ROWS_, D_, DFFP_);
}
